// RNN_42915313221650
// MI455X (gfx1250) — matrix operand order repaired and checked by reading, not compiled, not run on hardware
//
#include <hip/hip_runtime.h>

// ---------------------------------------------------------------------------
// Elman RNN on MI455X (gfx1250): bf16 WMMA 16x16x32, f32 accumulate.
//  Phase 1: xW = x @ W_ih^T + bias  (32768x1024x1024 GEMM, unroll-2 pipelined)
//  Phase 2: single persistent kernel, 32 WGs, each owns a 32-column n-slice;
//           its W_hh slice (64 KB) lives in LDS for all 512 steps; steps are
//           separated by a device-wide atomic barrier. h ping-pongs in global.
// ---------------------------------------------------------------------------

typedef __bf16 bf16_t;
typedef __attribute__((ext_vector_type(2)))  __bf16 v2bf;
typedef __attribute__((ext_vector_type(8)))  __bf16 v8bf;
typedef __attribute__((ext_vector_type(16))) __bf16 v16bf;
typedef __attribute__((ext_vector_type(8)))  float  v8f;
typedef __attribute__((ext_vector_type(4)))  float  v4f;
typedef __attribute__((ext_vector_type(8)))  unsigned vu8;

constexpr int SEQ  = 512;
constexpr int BATT = 64;
constexpr int KDIM = 1024;
constexpr int HDIM = 1024;
constexpr int BH   = BATT * HDIM;      // 65536
constexpr int NWG2 = 32;               // phase-2 persistent workgroups
constexpr int LDS_ROW = 1032;          // halves per W row in LDS (2064 B: 16B-aligned,
                                       // (2064/4)%64 = 4 -> B-frag lanes hit distinct banks)

// scalar round-to-nearest-even f32 -> bf16 (cold paths only)
__device__ __forceinline__ bf16_t f2bf(float f) {
    unsigned u = __builtin_bit_cast(unsigned, f);
    unsigned r = (u + 0x7FFFu + ((u >> 16) & 1u)) >> 16;
    unsigned short h = (unsigned short)r;
    return __builtin_bit_cast(bf16_t, h);
}

// pack two f32 into one dword of bf16 pair (hot path of phase 1)
__device__ __forceinline__ unsigned pkbf(float x, float y) {
#if __has_builtin(__builtin_amdgcn_cvt_pk_bf16_f32)
    v2bf p = __builtin_amdgcn_cvt_pk_bf16_f32(x, y);
    return __builtin_bit_cast(unsigned, p);
#else
    // round-half-up + byte-perm: 3 VALU per 2 elements
    unsigned ux = __builtin_bit_cast(unsigned, x) + 0x8000u;
    unsigned uy = __builtin_bit_cast(unsigned, y) + 0x8000u;
    return __builtin_amdgcn_perm(uy, ux, 0x07060302u);  // {uy[31:16], ux[31:16]}
#endif
}

__device__ __forceinline__ v16bf concat(v8bf lo, v8bf hi) {
    return __builtin_shufflevector(lo, hi, 0, 1, 2, 3, 4, 5, 6, 7,
                                           8, 9, 10, 11, 12, 13, 14, 15);
}

// A fragment (16x32 bf16), lane m = lane%16, g = lane/16; f32 source, cvt on the fly.
__device__ __forceinline__ v16bf load_a_f32(const float* __restrict__ row, int k0, int g) {
    const v4f* p0 = (const v4f*)(row + k0 + g * 8);
    const v4f* p1 = (const v4f*)(row + k0 + 16 + g * 8);
    v4f f0 = p0[0], f1 = p0[1], f2 = p1[0], f3 = p1[1];
    vu8 u;
    u[0] = pkbf(f0[0], f0[1]); u[1] = pkbf(f0[2], f0[3]);
    u[2] = pkbf(f1[0], f1[1]); u[3] = pkbf(f1[2], f1[3]);
    u[4] = pkbf(f2[0], f2[1]); u[5] = pkbf(f2[2], f2[3]);
    u[6] = pkbf(f3[0], f3[1]); u[7] = pkbf(f3[2], f3[3]);
    return __builtin_bit_cast(v16bf, u);
}

// A fragment from a bf16 row (global).
__device__ __forceinline__ v16bf load_a_bf16(const bf16_t* __restrict__ row, int k0, int g) {
    v8bf lo = *(const v8bf*)(row + k0 + g * 8);
    v8bf hi = *(const v8bf*)(row + k0 + 16 + g * 8);
    return concat(lo, hi);
}

// B fragment (32x16 bf16): lane n = lane%16 reads W row n, K k0+8g..+7 and k0+16+8g..+7 (global).
__device__ __forceinline__ v16bf load_b(const bf16_t* __restrict__ row, int k0, int g) {
    v8bf lo = *(const v8bf*)(row + k0 + g * 8);
    v8bf hi = *(const v8bf*)(row + k0 + 16 + g * 8);
    return concat(lo, hi);
}

// B fragment from LDS (padded row pitch).
__device__ __forceinline__ v16bf load_b_lds(const bf16_t* row, int k0, int g) {
    v8bf lo = *(const v8bf*)(row + k0 + g * 8);
    v8bf hi = *(const v8bf*)(row + k0 + 16 + g * 8);
    return concat(lo, hi);
}

// ---------------- utility kernels ----------------

__global__ __launch_bounds__(256) void k_f32_to_bf16(const float* __restrict__ in,
                                                     bf16_t* __restrict__ out, int n) {
    int i = blockIdx.x * 256 + threadIdx.x;
    if (i < n) out[i] = f2bf(in[i]);
}

__global__ __launch_bounds__(256) void k_init(bf16_t* __restrict__ h0, int* __restrict__ ctr) {
    int i = blockIdx.x * 256 + threadIdx.x;
    if (i < BH) h0[i] = f2bf(0.0f);
    if (i == 0) *ctr = 0;
}

__global__ __launch_bounds__(256) void k_copy_f32(const float* __restrict__ src,
                                                  float* __restrict__ dst, int n) {
    int i = blockIdx.x * 256 + threadIdx.x;
    if (i < n) dst[i] = src[i];
}

// ---------------- Phase 1: input GEMM (unroll-2 software pipeline) ----------------
// 8 waves/block, each wave: 1 m-tile x 4 n-tiles; grid 4096 blocks = 32768 waves.
__global__ __launch_bounds__(256) void k_gemm_xw(const float*  __restrict__ x,
                                                 const bf16_t* __restrict__ Wih,
                                                 const float*  __restrict__ b_ih,
                                                 const float*  __restrict__ b_hh,
                                                 float*        __restrict__ out) {
    const int lane = threadIdx.x & 31;
    const int wave = threadIdx.x >> 5;
    const int w    = blockIdx.x * 8 + wave;
    const int m0   = (w >> 4) * 16;
    const int n0   = (w & 15) * 64;
    const int g    = lane >> 4;
    const int ln   = lane & 15;

    const float* arow = x + (size_t)(m0 + ln) * KDIM;
    const bf16_t* brow[4];
#pragma unroll
    for (int j = 0; j < 4; ++j) brow[j] = Wih + (size_t)(n0 + j * 16 + ln) * KDIM;

    v8f acc[4] = {};

    // prologue: fill buffer set 0 with k=0
    v16bf a0 = load_a_f32(arow, 0, g);
    v16bf b0[4], b1[4];
#pragma unroll
    for (int j = 0; j < 4; ++j) b0[j] = load_b(brow[j], 0, g);
    v16bf a1;

    for (int k0 = 0; k0 < KDIM; k0 += 64) {
        // prefetch k0+32 into set 1, consume set 0
        a1 = load_a_f32(arow, k0 + 32, g);
#pragma unroll
        for (int j = 0; j < 4; ++j) b1[j] = load_b(brow[j], k0 + 32, g);
#pragma unroll
        for (int j = 0; j < 4; ++j)
            acc[j] = __builtin_amdgcn_wmma_f32_16x16x32_bf16(
                false, a0, false, b0[j], (short)0, acc[j], false, false);

        // prefetch (k0+64) into set 0 (wraps to 0 on last iter; harmless), consume set 1
        const int kn = (k0 + 64) & (KDIM - 1);
        a0 = load_a_f32(arow, kn, g);
#pragma unroll
        for (int j = 0; j < 4; ++j) b0[j] = load_b(brow[j], kn, g);
#pragma unroll
        for (int j = 0; j < 4; ++j)
            acc[j] = __builtin_amdgcn_wmma_f32_16x16x32_bf16(
                false, a1, false, b1[j], (short)0, acc[j], false, false);
    }

#pragma unroll
    for (int j = 0; j < 4; ++j) {
        const int col  = n0 + j * 16 + ln;
        const float bv = b_ih[col] + b_hh[col];
#pragma unroll
        for (int r = 0; r < 8; ++r) {
            const int row = m0 + g * 8 + r;
            out[(size_t)row * HDIM + col] = acc[j][r] + bv;
        }
    }
}

// ---------------- Phase 2: persistent recurrent scan ----------------
// 32 WGs x 256 threads. WG wg owns output cols [wg*32, wg*32+32): its W_hh
// slice (32 rows x 1024) is staged once into LDS (padded pitch, conflict-free)
// and reused for all 512 steps. Each of the 8 waves computes one 16x16 tile.
// h ping-pongs between two global bf16 buffers; steps separated by a
// device-wide monotonic atomic-counter barrier (32 WGs trivially co-resident).
__global__ __launch_bounds__(256) void k_rnn_scan(const bf16_t* __restrict__ Whh,
                                                  float*        __restrict__ out,
                                                  bf16_t*       __restrict__ h0,
                                                  bf16_t*       __restrict__ h1,
                                                  int*          __restrict__ ctr) {
    __shared__ bf16_t sW[32 * LDS_ROW];          // 66 KB of 320 KB

    const int wg   = blockIdx.x;
    const int tid  = threadIdx.x;
    const int lane = tid & 31;
    const int wv   = tid >> 5;
    const int mt   = wv & 3;                     // m-tile (batch rows)
    const int jt   = wv >> 2;                    // n-subtile within the 32-col slice
    const int g    = lane >> 4;
    const int ln   = lane & 15;
    const int m0   = mt * 16;
    const int col  = wg * 32 + jt * 16 + ln;

    // One-time stage of this WG's 64 KB W_hh slice into LDS.
    for (int i = tid; i < 32 * 128; i += 256) {
        const int r = i >> 7;
        const int c = (i & 127) * 8;
        *(uint4*)(sW + r * LDS_ROW + c) =
            *(const uint4*)(Whh + (size_t)(wg * 32 + r) * KDIM + c);
    }
    __syncthreads();

    const bf16_t* bbase = sW + (jt * 16 + ln) * LDS_ROW;

    for (int s = 0; s < SEQ; ++s) {
        const bf16_t* hp = (s & 1) ? h1 : h0;
        bf16_t*       hn = (s & 1) ? h0 : h1;
        float* outs = out + (size_t)s * BH;
        const bf16_t* arow = hp + (size_t)(m0 + ln) * HDIM;

        // Prefetch epilogue xW values; independent of the K loop -> fully hidden.
        float xw[8];
#pragma unroll
        for (int r = 0; r < 8; ++r)
            xw[r] = outs[(size_t)(m0 + 8 * g + r) * HDIM + col];

        v8f acc = {};
        v16bf a0 = load_a_bf16(arow, 0, g);
        v16bf b0 = load_b_lds(bbase, 0, g);
        for (int k0 = 0; k0 < KDIM; k0 += 64) {
            v16bf a1 = load_a_bf16(arow, k0 + 32, g);
            v16bf b1 = load_b_lds(bbase, k0 + 32, g);
            acc = __builtin_amdgcn_wmma_f32_16x16x32_bf16(
                false, a0, false, b0, (short)0, acc, false, false);
            const int kn = (k0 + 64) & (KDIM - 1);
            a0 = load_a_bf16(arow, kn, g);
            b0 = load_b_lds(bbase, kn, g);
            acc = __builtin_amdgcn_wmma_f32_16x16x32_bf16(
                false, a1, false, b1, (short)0, acc, false, false);
        }

#pragma unroll
        for (int r = 0; r < 8; ++r) {
            const int row = m0 + 8 * g + r;
            const size_t idx = (size_t)row * HDIM + col;
            const float h = tanhf(xw[r] + acc[r]);
            outs[idx] = h;
            hn[idx]   = f2bf(h);
        }

        // Device-wide barrier: monotonic counter, release on arrive, acquire on spin.
        __syncthreads();
        if (tid == 0) {
            __hip_atomic_fetch_add(ctr, 1, __ATOMIC_RELEASE, __HIP_MEMORY_SCOPE_AGENT);
            const int tgt = NWG2 * (s + 1);
            while (__hip_atomic_load(ctr, __ATOMIC_ACQUIRE, __HIP_MEMORY_SCOPE_AGENT) < tgt)
                __builtin_amdgcn_s_sleep(1);
        }
        __syncthreads();
    }
}

// ---------------- launch ----------------

extern "C" void kernel_launch(void* const* d_in, const int* in_sizes, int n_in,
                              void* d_out, int out_size, void* d_ws, size_t ws_size,
                              hipStream_t stream) {
    const float* x    = (const float*)d_in[0];
    const float* W_ih = (const float*)d_in[1];
    const float* W_hh = (const float*)d_in[2];
    const float* b_ih = (const float*)d_in[3];
    const float* b_hh = (const float*)d_in[4];
    float* out = (float*)d_out;

    char* ws = (char*)d_ws;
    bf16_t* Wih_bf = (bf16_t*)ws;                                   // 2 MB
    bf16_t* Whh_bf = (bf16_t*)(ws + (size_t)2 * 1024 * 1024);       // 2 MB
    bf16_t* h0     = (bf16_t*)(ws + (size_t)4 * 1024 * 1024);       // 128 KB
    bf16_t* h1     = h0 + BH;                                       // 128 KB
    int*    ctr    = (int*)(ws + (size_t)4 * 1024 * 1024 + 2 * (size_t)BH * 2 + 256);

    k_f32_to_bf16<<<4096, 256, 0, stream>>>(W_ih, Wih_bf, KDIM * HDIM);
    k_f32_to_bf16<<<4096, 256, 0, stream>>>(W_hh, Whh_bf, KDIM * HDIM);
    k_init<<<BH / 256, 256, 0, stream>>>(h0, ctr);

    k_gemm_xw<<<4096, 256, 0, stream>>>(x, Wih_bf, b_ih, b_hh, out);

    k_rnn_scan<<<NWG2, 256, 0, stream>>>(Whh_bf, out, h0, h1, ctr);

    k_copy_f32<<<BH / 256, 256, 0, stream>>>(out + (size_t)(SEQ - 1) * BH,
                                             out + (size_t)SEQ * BH, BH);
}
